// SelfAttention_14345190769329
// MI455X (gfx1250) — matrix operand order repaired and checked by reading, not compiled, not run on hardware
//
#include <hip/hip_runtime.h>
#include <hip/hip_bf16.h>

typedef __attribute__((ext_vector_type(16))) __bf16 v16bf;
typedef __attribute__((ext_vector_type(8)))  __bf16 v8bf;
typedef __attribute__((ext_vector_type(4)))  __bf16 v4bf;
typedef __attribute__((ext_vector_type(8)))  float  v8f;

#define DIM   768
#define NHEAD 12
#define HDIM  64
#define ATT_SCALE 0.125f  /* 64^-0.5 */

// Optional gfx1250 async global->LDS copy path (ASYNCcnt).
#if defined(__has_builtin)
#  if __has_builtin(__builtin_amdgcn_global_load_async_to_lds_b128) && \
      __has_builtin(__builtin_amdgcn_s_wait_asynccnt)
#    define HAVE_ASYNC_LDS 1
#  endif
#endif
#ifndef HAVE_ASYNC_LDS
#  define HAVE_ASYNC_LDS 0
#endif
#if HAVE_ASYNC_LDS
typedef int int4v __attribute__((ext_vector_type(4)));
typedef __attribute__((address_space(1))) int4v glb_i4;   // global (device) AS
typedef __attribute__((address_space(3))) int4v lds_i4;   // LDS AS
#endif

// K index this code's A/B fragments hold in VGPR j for lane-group g (contiguous k order):
// lanes 0-15 (g=0): VGPR0-7 -> K 0..15
// lanes16-31 (g=1): VGPR0-7 -> K 16..31
__device__ __forceinline__ int a_kidx(int j, int g) {
    return g * 16 + j * 2;
}

// ---------------------------------------------------------------------------
// Kernel 1: qkv = x @ w_qkv ; scatter to Q[H][L][64] (*SCALE), K[H][L][64],
//           V^T[H][64][L], all bf16.  WG tile 128M x 64N, 8 waves.
//           Software-pipelined: next K-slab prefetched into VGPRs during WMMA.
// ---------------------------------------------------------------------------
__global__ __launch_bounds__(256)
void qkv_gemm(const float* __restrict__ x, const float* __restrict__ w,
              __bf16* __restrict__ q, __bf16* __restrict__ k,
              __bf16* __restrict__ vT, int L)
{
    __shared__ __bf16 la[128][36];   // A tile, row-major [m][k]
    __shared__ __bf16 lb[64][36];    // B tile, transposed [n][k]

    const int tid  = threadIdx.x;
    const int wv   = tid >> 5, lane = tid & 31;
    const int g    = lane >> 4, hl  = lane & 15;
    const int m0   = blockIdx.y * 128;
    const int n0   = blockIdx.x * 64;
    const int N    = 3 * DIM;

    float4 ax[4], bx[2];
    auto loadreg = [&](int k0) {
        #pragma unroll
        for (int i = 0; i < 4; ++i) {
            int fid = i * 256 + tid;
            int r = fid >> 3, c4 = (fid & 7) << 2;
            ax[i] = *(const float4*)(x + (size_t)(m0 + r) * DIM + k0 + c4);
        }
        #pragma unroll
        for (int i = 0; i < 2; ++i) {
            int fid = i * 256 + tid;
            int kr = fid >> 4, nc4 = (fid & 15) << 2;
            bx[i] = *(const float4*)(w + (size_t)(k0 + kr) * N + n0 + nc4);
        }
    };
    auto store_lds = [&]() {
        #pragma unroll
        for (int i = 0; i < 4; ++i) {
            int fid = i * 256 + tid;
            int r = fid >> 3, c4 = (fid & 7) << 2;
            v4bf h4 = { (__bf16)ax[i].x, (__bf16)ax[i].y,
                        (__bf16)ax[i].z, (__bf16)ax[i].w };
            *(v4bf*)&la[r][c4] = h4;
        }
        #pragma unroll
        for (int i = 0; i < 2; ++i) {
            int fid = i * 256 + tid;
            int kr = fid >> 4, nc4 = (fid & 15) << 2;
            lb[nc4 + 0][kr] = (__bf16)bx[i].x;
            lb[nc4 + 1][kr] = (__bf16)bx[i].y;
            lb[nc4 + 2][kr] = (__bf16)bx[i].z;
            lb[nc4 + 3][kr] = (__bf16)bx[i].w;
        }
    };

    v8f acc[4] = {};
    loadreg(0);

    for (int k0 = 0; k0 < DIM; k0 += 32) {
        store_lds();
        __syncthreads();
        if (k0 + 32 < DIM) loadreg(k0 + 32);   // overlap with WMMA below

        v16bf af;
        #pragma unroll
        for (int j = 0; j < 8; ++j) {
            int kk = a_kidx(j, g);
            af[2 * j]     = la[wv * 16 + hl][kk];
            af[2 * j + 1] = la[wv * 16 + hl][kk + 1];
        }
        #pragma unroll
        for (int nt = 0; nt < 4; ++nt) {
            v16bf bfr;
            const int kb = g * 16;
            #pragma unroll
            for (int j = 0; j < 8; ++j) {
                bfr[2 * j]     = lb[nt * 16 + hl][kb + 2 * j];
                bfr[2 * j + 1] = lb[nt * 16 + hl][kb + 2 * j + 1];
            }
            acc[nt] = __builtin_amdgcn_wmma_f32_16x16x32_bf16(
                false, af, false, bfr, (short)0, acc[nt], false, false);
        }
        __syncthreads();
    }

    // Epilogue: 64-wide N blocks align to one (section, head) -> scalar route.
    const int sec = n0 / DIM;
    const int c0  = n0 % DIM;
    const int hh  = c0 / HDIM;

    if (sec < 2) {
        __bf16* dst = (sec == 0) ? q : k;
        const float scl = (sec == 0) ? ATT_SCALE : 1.0f;
        __bf16* base = dst + (size_t)hh * L * HDIM;
        #pragma unroll
        for (int nt = 0; nt < 4; ++nt)
            #pragma unroll
            for (int r = 0; r < 8; ++r) {
                int mg = m0 + wv * 16 + r + 8 * g;
                int d  = nt * 16 + hl;
                base[(size_t)mg * HDIM + d] = (__bf16)(acc[nt][r] * scl);
            }
    } else {
        __bf16* base = vT + (size_t)hh * HDIM * L;
        #pragma unroll
        for (int nt = 0; nt < 4; ++nt)
            #pragma unroll
            for (int r = 0; r < 8; ++r) {
                int mg = m0 + wv * 16 + r + 8 * g;
                int d  = nt * 16 + hl;
                base[(size_t)d * L + mg] = (__bf16)acc[nt][r];
            }
    }
}

// ---------------------------------------------------------------------------
// Kernel 2: flash attention.  grid = (L/128, H), 8 waves, 16 queries/wave.
// Double-buffered K/V tiles; next tile staged (async-to-LDS if available)
// while WMMAs run on the current one.  One barrier per 32-key step.
// ---------------------------------------------------------------------------
__global__ __launch_bounds__(256)
void attn_kernel(const __bf16* __restrict__ qm, const __bf16* __restrict__ km,
                 const __bf16* __restrict__ vT, __bf16* __restrict__ attn, int L)
{
    __shared__ __bf16 kt[2][32][72];   // [buf][key][d]  (144B rows, 16B aligned)
    __shared__ __bf16 vt[2][64][40];   // [buf][d][key]  ( 80B rows, 16B aligned)
    __shared__ float  ls[8][16][36];   // per-wave raw scores (16B-aligned halves)
    __shared__ __bf16 lp[8][16][32];   // per-wave exp(P), A-fragment source
    __shared__ float  lpm[8][16][2];   // partial row max
    __shared__ float  lps[8][16][2];   // partial row sum
    __shared__ float  lstat[8][16];    // alpha / 1/l broadcast

    const int tid  = threadIdx.x;
    const int wv   = tid >> 5, lane = tid & 31;
    const int g    = lane >> 4, hl  = lane & 15;
    const int h    = blockIdx.y;
    const int q0   = blockIdx.x * 128 + wv * 16;

    const __bf16* qh = qm + (size_t)h * L * HDIM;
    const __bf16* kh = km + (size_t)h * L * HDIM;
    const __bf16* vh = vT + (size_t)h * HDIM * L;

    auto stage = [&](int b, int key0) {
        const int krow = tid >> 3, kc8 = (tid & 7) << 3;
        const int vrow = tid >> 2, vc8 = (tid & 3) << 3;
        const __bf16* gk = kh + (size_t)(key0 + krow) * HDIM + kc8;
        const __bf16* gv = vh + (size_t)vrow * L + key0 + vc8;
#if HAVE_ASYNC_LDS
        __builtin_amdgcn_global_load_async_to_lds_b128(
            (glb_i4*)(uintptr_t)gk,
            (lds_i4*)(uint32_t)(uintptr_t)&kt[b][krow][kc8], 0, 0);
        __builtin_amdgcn_global_load_async_to_lds_b128(
            (glb_i4*)(uintptr_t)gv,
            (lds_i4*)(uint32_t)(uintptr_t)&vt[b][vrow][vc8], 0, 0);
#else
        *(v8bf*)&kt[b][krow][kc8] = *(const v8bf*)gk;
        *(v8bf*)&vt[b][vrow][vc8] = *(const v8bf*)gv;
#endif
    };

    v16bf aq[2];                       // Q fragments, 2 chunks over d (load once)
    #pragma unroll
    for (int c = 0; c < 2; ++c)
        #pragma unroll
        for (int j = 0; j < 8; ++j) {
            int kk = c * 32 + a_kidx(j, g);
            aq[c][2 * j]     = qh[(size_t)(q0 + hl) * HDIM + kk];
            aq[c][2 * j + 1] = qh[(size_t)(q0 + hl) * HDIM + kk + 1];
        }

    v8f o[4] = {};
    float mrun = -3.0e38f, lrun = 0.f;

    stage(0, 0);
    for (int key0 = 0; key0 < L; key0 += 32) {
        const int cur = (key0 >> 5) & 1;
#if HAVE_ASYNC_LDS
        __builtin_amdgcn_s_wait_asynccnt(0);
#endif
        __syncthreads();               // buf[cur] ready; buf[cur^1] reads done
        if (key0 + 32 < L) stage(cur ^ 1, key0 + 32);   // overlaps WMMAs below

        // S = Q K^T over 32 keys: two 16x16 tiles, 2 WMMAs each
        v8f s[2] = {};
        #pragma unroll
        for (int t = 0; t < 2; ++t) {
            #pragma unroll
            for (int c = 0; c < 2; ++c) {
                v16bf bk;
                const int kb = c * 32 + g * 16;
                #pragma unroll
                for (int j = 0; j < 8; ++j) {
                    bk[2 * j]     = kt[cur][t * 16 + hl][kb + 2 * j];
                    bk[2 * j + 1] = kt[cur][t * 16 + hl][kb + 2 * j + 1];
                }
                s[t] = __builtin_amdgcn_wmma_f32_16x16x32_bf16(
                    false, aq[c], false, bk, (short)0, s[t], false, false);
            }
        }
        // scatter S to per-wave LDS (wave-internal; DScnt ordering suffices)
        #pragma unroll
        for (int t = 0; t < 2; ++t)
            #pragma unroll
            for (int r = 0; r < 8; ++r)
                ls[wv][r + 8 * g][t * 16 + hl] = s[t][r];

        // online softmax: 2 lanes per query row (row = hl, half = g)
        {
            const int row = hl, half = g, cb = half * 16;
            float pm = -3.0e38f;
            #pragma unroll
            for (int j2 = 0; j2 < 16; ++j2) pm = fmaxf(pm, ls[wv][row][cb + j2]);
            lpm[wv][row][half] = pm;
            float mx = fmaxf(mrun, fmaxf(lpm[wv][row][0], lpm[wv][row][1]));
            float alpha = __expf(mrun - mx);
            float psum = 0.f;
            #pragma unroll
            for (int j2 = 0; j2 < 16; ++j2) {
                float e = __expf(ls[wv][row][cb + j2] - mx);
                lp[wv][row][cb + j2] = (__bf16)e;
                psum += e;
            }
            lps[wv][row][half] = psum;
            lrun = lrun * alpha + lps[wv][row][0] + lps[wv][row][1];
            mrun = mx;
            if (half == 0) lstat[wv][row] = alpha;
        }

        // rescale O by alpha(row)
        #pragma unroll
        for (int r = 0; r < 8; ++r) {
            float a = lstat[wv][r + 8 * g];
            #pragma unroll
            for (int nt = 0; nt < 4; ++nt) o[nt][r] *= a;
        }

        // P fragment from LDS
        v16bf pf;
        #pragma unroll
        for (int j = 0; j < 8; ++j) {
            int kk = a_kidx(j, g);
            pf[2 * j]     = lp[wv][hl][kk];
            pf[2 * j + 1] = lp[wv][hl][kk + 1];
        }
        // O += P * V
        #pragma unroll
        for (int nt = 0; nt < 4; ++nt) {
            v16bf bv;
            const int d  = nt * 16 + hl;
            const int kb = g * 16;
            #pragma unroll
            for (int j = 0; j < 8; ++j) {
                bv[2 * j]     = vt[cur][d][kb + 2 * j];
                bv[2 * j + 1] = vt[cur][d][kb + 2 * j + 1];
            }
            o[nt] = __builtin_amdgcn_wmma_f32_16x16x32_bf16(
                false, pf, false, bv, (short)0, o[nt], false, false);
        }
    }

    // final 1/l normalize + store (per-wave LDS, wave-internal ordering)
    if (g == 0) lstat[wv][hl] = (lrun > 0.f) ? (1.f / lrun) : 0.f;
    #pragma unroll
    for (int r = 0; r < 8; ++r) {
        float inv = lstat[wv][r + 8 * g];
        int   mg  = q0 + r + 8 * g;
        #pragma unroll
        for (int nt = 0; nt < 4; ++nt) {
            int col = h * HDIM + nt * 16 + hl;
            attn[(size_t)mg * DIM + col] = (__bf16)(o[nt][r] * inv);
        }
    }
}

// ---------------------------------------------------------------------------
// Kernel 3: out = attn(bf16) @ w_proj + b_proj, fp32 output.  Pipelined.
// ---------------------------------------------------------------------------
__global__ __launch_bounds__(256)
void proj_gemm(const __bf16* __restrict__ a, const float* __restrict__ w,
               const float* __restrict__ bias, float* __restrict__ out, int L)
{
    __shared__ __bf16 la[128][36];
    __shared__ __bf16 lb[64][36];

    const int tid  = threadIdx.x;
    const int wv   = tid >> 5, lane = tid & 31;
    const int g    = lane >> 4, hl  = lane & 15;
    const int m0   = blockIdx.y * 128;
    const int n0   = blockIdx.x * 64;

    v8bf  ar[2];
    float4 bx[2];
    auto loadreg = [&](int k0) {
        #pragma unroll
        for (int i = 0; i < 2; ++i) {
            int fid = i * 256 + tid;
            int r = fid >> 2, c8 = (fid & 3) << 3;
            ar[i] = *(const v8bf*)(a + (size_t)(m0 + r) * DIM + k0 + c8);
        }
        #pragma unroll
        for (int i = 0; i < 2; ++i) {
            int fid = i * 256 + tid;
            int kr = fid >> 4, nc4 = (fid & 15) << 2;
            bx[i] = *(const float4*)(w + (size_t)(k0 + kr) * DIM + n0 + nc4);
        }
    };
    auto store_lds = [&]() {
        #pragma unroll
        for (int i = 0; i < 2; ++i) {
            int fid = i * 256 + tid;
            int r = fid >> 2, c8 = (fid & 3) << 3;
            *(v8bf*)&la[r][c8] = ar[i];
        }
        #pragma unroll
        for (int i = 0; i < 2; ++i) {
            int fid = i * 256 + tid;
            int kr = fid >> 4, nc4 = (fid & 15) << 2;
            lb[nc4 + 0][kr] = (__bf16)bx[i].x;
            lb[nc4 + 1][kr] = (__bf16)bx[i].y;
            lb[nc4 + 2][kr] = (__bf16)bx[i].z;
            lb[nc4 + 3][kr] = (__bf16)bx[i].w;
        }
    };

    v8f acc[4] = {};
    loadreg(0);

    for (int k0 = 0; k0 < DIM; k0 += 32) {
        store_lds();
        __syncthreads();
        if (k0 + 32 < DIM) loadreg(k0 + 32);   // overlap with WMMA below

        v16bf af;
        #pragma unroll
        for (int j = 0; j < 8; ++j) {
            int kk = a_kidx(j, g);
            af[2 * j]     = la[wv * 16 + hl][kk];
            af[2 * j + 1] = la[wv * 16 + hl][kk + 1];
        }
        #pragma unroll
        for (int nt = 0; nt < 4; ++nt) {
            v16bf bfr;
            const int kb = g * 16;
            #pragma unroll
            for (int j = 0; j < 8; ++j) {
                bfr[2 * j]     = lb[nt * 16 + hl][kb + 2 * j];
                bfr[2 * j + 1] = lb[nt * 16 + hl][kb + 2 * j + 1];
            }
            acc[nt] = __builtin_amdgcn_wmma_f32_16x16x32_bf16(
                false, af, false, bfr, (short)0, acc[nt], false, false);
        }
        __syncthreads();
    }

    #pragma unroll
    for (int nt = 0; nt < 4; ++nt) {
        #pragma unroll
        for (int r = 0; r < 8; ++r) {
            int mg = m0 + wv * 16 + r + 8 * g;
            int ng = n0 + nt * 16 + hl;
            out[(size_t)mg * DIM + ng] = acc[nt][r] + bias[ng];
        }
    }
}

// ---------------------------------------------------------------------------
extern "C" void kernel_launch(void* const* d_in, const int* in_sizes, int n_in,
                              void* d_out, int out_size, void* d_ws, size_t ws_size,
                              hipStream_t stream)
{
    const float* x      = (const float*)d_in[0];
    const float* w_qkv  = (const float*)d_in[1];
    const float* w_proj = (const float*)d_in[2];
    const float* b_proj = (const float*)d_in[3];
    float*       out    = (float*)d_out;

    const int L = in_sizes[0] / DIM;   // 4096

    char* ws = (char*)d_ws;
    const size_t headBytes = (size_t)NHEAD * (size_t)L * HDIM * sizeof(__bf16);
    __bf16* q    = (__bf16*)(ws);
    __bf16* k    = (__bf16*)(ws + headBytes);
    __bf16* vT   = (__bf16*)(ws + 2 * headBytes);
    __bf16* attn = (__bf16*)(ws + 3 * headBytes);

    dim3 blk(256);
    qkv_gemm  <<<dim3(3 * DIM / 64, L / 128), blk, 0, stream>>>(x, w_qkv, q, k, vT, L);
    attn_kernel<<<dim3(L / 128, NHEAD),       blk, 0, stream>>>(q, k, vT, attn, L);
    proj_gemm <<<dim3(DIM / 64, L / 128),     blk, 0, stream>>>(attn, w_proj, b_proj, out, L);
}
